// WanAttentionBlock_81106162417699
// MI455X (gfx1250) — matrix operand order repaired and checked by reading, not compiled, not run on hardware
//
#include <hip/hip_runtime.h>

// ---------------- problem constants ----------------
#define DIMC 1536
#define NHC  12
#define HDC  128
#define HCC  64
#define FFNC 8960
#define LQC  2048
#define LKXC 512
#define EPSC 1e-6f
#define SCALEC 0.08838834764831845f   // 1/sqrt(128)

typedef __attribute__((ext_vector_type(16))) __bf16 v16bf;
typedef __attribute__((ext_vector_type(8)))  __bf16 v8bf;
typedef __attribute__((ext_vector_type(8)))  float  v8f;

#if __has_builtin(__builtin_amdgcn_tensor_load_to_lds)
#define USE_TDM 1
#pragma message("CDNA5: tensor_load_to_lds builtin AVAILABLE -> TDM staging enabled")
#else
#define USE_TDM 0
#pragma message("CDNA5: tensor_load_to_lds builtin NOT available -> cooperative-copy fallback")
#endif
#if __has_builtin(__builtin_amdgcn_cvt_pk_bf16_f32)
#pragma message("CDNA5: cvt_pk_bf16_f32 builtin AVAILABLE")
#else
#pragma message("CDNA5: cvt_pk_bf16_f32 builtin NOT available -> perm pack fallback")
#endif

// ---------------- scalar/packed fp32 -> bf16 ----------------
__device__ inline unsigned pk2(float a, float b) {
#if __has_builtin(__builtin_amdgcn_cvt_pk_bf16_f32)
  auto t = __builtin_amdgcn_cvt_pk_bf16_f32(a, b);
  return __builtin_bit_cast(unsigned, t);
#else
  unsigned ua = __builtin_bit_cast(unsigned, a) + 0x8000u;
  unsigned ub = __builtin_bit_cast(unsigned, b) + 0x8000u;
  return __builtin_amdgcn_perm(ub, ua, 0x07060302u); // {hi16(b), hi16(a)}
#endif
}
__device__ inline __bf16 f2bf(float f) {
  unsigned u = __builtin_bit_cast(unsigned, f) + 0x8000u;
  return __builtin_bit_cast(__bf16, (unsigned short)(u >> 16));
}
__device__ inline float bf2f(__bf16 b) {
  unsigned u = (unsigned)__builtin_bit_cast(unsigned short, b) << 16;
  return __builtin_bit_cast(float, u);
}

__device__ inline v8f wmma_bf16(v16bf a, v16bf b, v8f c) {
  return __builtin_amdgcn_wmma_f32_16x16x32_bf16(false, a, false, b, (short)0, c,
                                                 false, false);
}

// A-fragment (16x32 bf16): lane needs k = {0..7} and {16..23} relative to p. 2x16B loads.
__device__ inline v16bf ldfragA(const __bf16* p) {
  const v8bf* q = (const v8bf*)p;
  v8bf a = q[0], b = q[2];
  v16bf r;
#pragma unroll
  for (int i = 0; i < 8; i++) { r[i] = a[i]; r[8 + i] = b[i]; }
  return r;
}
// B-fragment (32x16 bf16): lane needs k = {0..7} and {16..23} relative to p. 2x16B loads.
__device__ inline v16bf ldfragB(const __bf16* p) {
  const v8bf* q = (const v8bf*)p;
  v8bf a = q[0], b = q[2];
  v16bf r;
#pragma unroll
  for (int i = 0; i < 8; i++) { r[i] = a[i]; r[8 + i] = b[i]; }
  return r;
}

__device__ inline float gelu_tanh(float x) {
  float x3 = x * x * x;
  return 0.5f * x * (1.f + tanhf(0.7978845608028654f * (x + 0.044715f * x3)));
}

#if USE_TDM
typedef __attribute__((ext_vector_type(4))) unsigned tdm_g0_t;
typedef __attribute__((ext_vector_type(8))) int      tdm_g1_t;
typedef __attribute__((ext_vector_type(4))) int      tdm_g2_t;

// 2D tile load: tile (tile0 x tile1) of bf16 from row-major [dim1][dim0] tensor
// (row stride = stride0 elements) starting at gsrc, into contiguous LDS at lds_off.
__device__ inline void tdm_load_2d(const __bf16* gsrc, unsigned lds_off,
                                   int dim0, int dim1, int stride0,
                                   int tile0, int tile1) {
  unsigned long long ga = (unsigned long long)(uintptr_t)gsrc;
  tdm_g0_t g0;
  g0[0] = 1u;                                               // count=1 (valid user D#)
  g0[1] = lds_off;                                          // lds_addr
  g0[2] = (unsigned)(ga & 0xFFFFFFFFu);                     // global_addr[31:0]
  g0[3] = (unsigned)((ga >> 32) & 0x1FFFFFFu) | (2u << 30); // addr[56:32] | type=2
  tdm_g1_t g1;
  g1[0] = (1 << 16);                                        // data_size=1 (2 bytes)
  g1[1] = (int)(((unsigned)dim0 & 0xFFFFu) << 16);          // tensor_dim0[15:0]
  g1[2] = (int)((((unsigned)dim0 >> 16) & 0xFFFFu) |
                (((unsigned)dim1 & 0xFFFFu) << 16));        // dim0[31:16] | dim1[15:0]
  g1[3] = (int)((((unsigned)dim1 >> 16) & 0xFFFFu) |
                ((unsigned)tile0 << 16));                   // dim1[31:16] | tile_dim0
  g1[4] = tile1 & 0xFFFF;                                   // tile_dim1 (tile_dim2=0)
  g1[5] = stride0;                                          // tensor_dim0_stride[31:0]
  g1[6] = 0;
  g1[7] = 0;
  tdm_g2_t gz = {0, 0, 0, 0};
#if __clang_major__ >= 23
  tdm_g1_t gz8 = {0, 0, 0, 0, 0, 0, 0, 0};
  __builtin_amdgcn_tensor_load_to_lds(g0, g1, gz, gz, gz8, 0);
#else
  __builtin_amdgcn_tensor_load_to_lds(g0, g1, gz, gz, 0);
#endif
}
#endif

// ---------------- fp32 -> bf16 streaming convert (8 elems/thread) ----------------
__global__ __launch_bounds__(256) void cvt_k(const float* __restrict__ src,
                                             __bf16* __restrict__ dst, int n8) {
  int i = blockIdx.x * 256 + threadIdx.x;
  if (i >= n8) return;
  const float4* s = (const float4*)src + 2 * (size_t)i;
  float4 a = s[0], b = s[1];
  uint4 o;
  o.x = pk2(a.x, a.y); o.y = pk2(a.z, a.w);
  o.z = pk2(b.x, b.y); o.w = pk2(b.z, b.w);
  ((uint4*)dst)[i] = o;
}

// ---------------- prep: EM = modulation + e ; rope cos/sin tables ----------------
__global__ __launch_bounds__(256) void prep_k(const float* __restrict__ e,
                                              const float* __restrict__ modulation,
                                              const float* __restrict__ freqs,
                                              float* __restrict__ EM,
                                              float* __restrict__ cosT,
                                              float* __restrict__ sinT) {
  int i = blockIdx.x * 256 + threadIdx.x;
  if (i < 6 * DIMC) EM[i] = modulation[i] + e[i];
  if (i < LQC * HCC) {
    int l = i >> 6, c = i & 63;
    int f = l >> 10;
    int hh = (l >> 5) & 31;
    int ww = l & 31;
    int p = (c < 22) ? f : ((c < 43) ? hh : ww);   // c1=22, c2=21
    float ang = freqs[p * HCC + c];
    cosT[i] = cosf(ang);
    sinT[i] = sinf(ang);
  }
}

// ---------------- layernorm (+modulation or affine), bf16 output ----------------
__global__ __launch_bounds__(256) void ln_k(const float* __restrict__ x,
                                            const float* __restrict__ w,
                                            const float* __restrict__ b,
                                            __bf16* __restrict__ out, int mode) {
  int row = blockIdx.x, tid = threadIdx.x;
  const float* xr = x + (size_t)row * DIMC;
  float v[6];
  float s = 0.f;
#pragma unroll
  for (int i = 0; i < 6; i++) { v[i] = xr[tid + 256 * i]; s += v[i]; }
  __shared__ float red[256];
  red[tid] = s; __syncthreads();
  for (int off = 128; off > 0; off >>= 1) {
    if (tid < off) red[tid] += red[tid + off];
    __syncthreads();
  }
  float mu = red[0] * (1.f / DIMC);
  __syncthreads();
  float vs = 0.f;
#pragma unroll
  for (int i = 0; i < 6; i++) { float d = v[i] - mu; vs += d * d; }
  red[tid] = vs; __syncthreads();
  for (int off = 128; off > 0; off >>= 1) {
    if (tid < off) red[tid] += red[tid + off];
    __syncthreads();
  }
  float inv = rsqrtf(red[0] * (1.f / DIMC) + EPSC);
#pragma unroll
  for (int i = 0; i < 6; i++) {
    int c = tid + 256 * i;
    float sc = mode ? w[c] : (1.f + w[c]);
    out[(size_t)row * DIMC + c] = f2bf((v[i] - mu) * inv * sc + b[c]);
  }
}

// ---------------- per-head RMS norm (+ optional RoPE), bf16 in/out ----------------
__global__ __launch_bounds__(128) void rmsrope_k(const __bf16* __restrict__ in,
                                                 const float* __restrict__ g,
                                                 const float* __restrict__ cosT,
                                                 const float* __restrict__ sinT,
                                                 __bf16* __restrict__ out, int rope) {
  int rh = blockIdx.x;
  int l = rh / NHC, h = rh % NHC;
  int d = threadIdx.x;
  size_t base = (size_t)l * DIMC + h * HDC;
  float v = bf2f(in[base + d]);
  __shared__ float red[128];
  __shared__ float sh[128];
  red[d] = v * v; __syncthreads();
  for (int off = 64; off > 0; off >>= 1) {
    if (d < off) red[d] += red[d + off];
    __syncthreads();
  }
  float inv = rsqrtf(red[0] * (1.f / HDC) + EPSC);
  float y = v * inv * g[h * HDC + d];
  if (rope) {
    sh[d] = y; __syncthreads();
    int c = d >> 1;
    float cs = cosT[(size_t)l * HCC + c], sn = sinT[(size_t)l * HCC + c];
    float yr = sh[c * 2], yi = sh[c * 2 + 1];
    y = (d & 1) ? (yr * sn + yi * cs) : (yr * cs - yi * sn);
  }
  out[base + d] = f2bf(y);
}

// ---------------- V transpose: V[l][c] -> VT[c][l]  (bf16 -> bf16) ----------------
__global__ __launch_bounds__(256) void transpose_k(const __bf16* __restrict__ V,
                                                   __bf16* __restrict__ VT, int Lk) {
  int i = blockIdx.x * 256 + threadIdx.x;
  if (i >= Lk * DIMC) return;
  int l = i / DIMC, c = i - l * DIMC;
  VT[(size_t)c * Lk + l] = V[i];
}

// ---------------- WMMA GEMM: C[M,N] = A[M,K] * W[N,K]^T (+ epilogue) --------------
enum { EPI_BF16 = 0, EPI_RESID_SCALE = 1, EPI_RESID = 2, EPI_GELU_BF16 = 3 };

template <int EPI>
__global__ __launch_bounds__(256) void gemm_k(const __bf16* __restrict__ A,
                                              const __bf16* __restrict__ W,
                                              const float* __restrict__ bias,
                                              const float* __restrict__ resid,
                                              const float* __restrict__ scale,
                                              float* __restrict__ outF,
                                              __bf16* __restrict__ outB,
                                              int M, int N, int K) {
  const int tid = threadIdx.x;
  const int lane = tid & 31, wv = tid >> 5;
  const int half = lane >> 4, l16 = lane & 15;
  const int wr = wv >> 2, wc = wv & 3;
  const int rowbase = blockIdx.y * 64;
  const int colbase = blockIdx.x * 256 + wc * 64;

  __shared__ __attribute__((aligned(128))) __bf16 Atile[2][64][32];

  v8f acc[2][4];
#pragma unroll
  for (int i = 0; i < 2; i++)
#pragma unroll
    for (int j = 0; j < 4; j++)
#pragma unroll
      for (int e = 0; e < 8; e++) acc[i][j][e] = 0.f;

  const __bf16* Ablk = A + (size_t)rowbase * K;
  const __bf16* wrow0 = W + (size_t)(colbase + 0 * 16 + l16) * K;
  const __bf16* wrow1 = W + (size_t)(colbase + 1 * 16 + l16) * K;
  const __bf16* wrow2 = W + (size_t)(colbase + 2 * 16 + l16) * K;
  const __bf16* wrow3 = W + (size_t)(colbase + 3 * 16 + l16) * K;

#if USE_TDM
  const unsigned lds_base = (unsigned)(uintptr_t)(&Atile[0][0][0]);
  if (wv == 0) tdm_load_2d(Ablk, lds_base, K, M, K, 32, 64);
  int parity = 0;
#endif

  for (int k0 = 0; k0 < K; k0 += 32) {
#if USE_TDM
    if (wv == 0) {
      if (k0 + 32 < K) {
        tdm_load_2d(Ablk + k0 + 32, lds_base + ((parity ^ 1) << 12), K, M, K, 32, 64);
        __builtin_amdgcn_s_wait_tensorcnt(1);
      } else {
        __builtin_amdgcn_s_wait_tensorcnt(0);
      }
    }
    __syncthreads();
    const __bf16* at = &Atile[parity][0][0];
#else
    __syncthreads();
    {
      int row = tid >> 2;
      int c16 = (tid & 3) * 8;
      *(uint4*)&Atile[0][row][c16] =
          *(const uint4*)(Ablk + (size_t)row * K + k0 + c16);
    }
    __syncthreads();
    const __bf16* at = &Atile[0][0][0];
#endif

    const int koa = half * 8;
    const int kob = k0 + half * 8;
    v16bf a0 = ldfragA(at + (wr * 32 + l16) * 32 + koa);
    v16bf a1 = ldfragA(at + (wr * 32 + 16 + l16) * 32 + koa);
    v16bf b0 = ldfragB(wrow0 + kob);
    v16bf b1 = ldfragB(wrow1 + kob);
    v16bf b2 = ldfragB(wrow2 + kob);
    v16bf b3 = ldfragB(wrow3 + kob);
    acc[0][0] = wmma_bf16(a0, b0, acc[0][0]);
    acc[1][0] = wmma_bf16(a1, b0, acc[1][0]);
    acc[0][1] = wmma_bf16(a0, b1, acc[0][1]);
    acc[1][1] = wmma_bf16(a1, b1, acc[1][1]);
    acc[0][2] = wmma_bf16(a0, b2, acc[0][2]);
    acc[1][2] = wmma_bf16(a1, b2, acc[1][2]);
    acc[0][3] = wmma_bf16(a0, b3, acc[0][3]);
    acc[1][3] = wmma_bf16(a1, b3, acc[1][3]);

    __syncthreads();
#if USE_TDM
    parity ^= 1;
#endif
  }

#pragma unroll
  for (int mi = 0; mi < 2; mi++)
#pragma unroll
    for (int t = 0; t < 4; t++)
#pragma unroll
      for (int r = 0; r < 8; r++) {
        int row = rowbase + wr * 32 + mi * 16 + r + 8 * half;
        int col = colbase + t * 16 + l16;
        size_t o = (size_t)row * N + col;
        float v = acc[mi][t][r] + bias[col];
        if (EPI == EPI_BF16)             outB[o] = f2bf(v);
        else if (EPI == EPI_RESID_SCALE) outF[o] = resid[o] + v * scale[col];
        else if (EPI == EPI_RESID)       outF[o] = resid[o] + v;
        else                             outB[o] = f2bf(gelu_tanh(v));
      }
}

// ---------------- flash attention: K/V^T tiles staged in LDS (TDM double-buffered) --
// Q,K layout [l][h*128+d] bf16 ; VT layout [h*128+d][Lk] bf16 ; Y layout [l][h*128+d]
__global__ __launch_bounds__(256) void attn_k(const __bf16* __restrict__ Q,
                                              const __bf16* __restrict__ K,
                                              const __bf16* __restrict__ VT,
                                              __bf16* __restrict__ Y,
                                              int Lk, float scale) {
  const int tid = threadIdx.x;
  const int lane = tid & 31, wv = tid >> 5;
  const int half = lane >> 4, l16 = lane & 15;
  const int h = blockIdx.y;
  const int qbase = blockIdx.x * 128 + wv * 16;

#if USE_TDM
#define NBUF 2
#else
#define NBUF 1
#endif
  __shared__ __attribute__((aligned(128))) __bf16 Ktile[NBUF][32][128];  // [key][d]
  __shared__ __attribute__((aligned(128))) __bf16 Vtile[NBUF][128][32];  // [d][key]
  __shared__ __bf16 P[8][16][32];

  v16bf qf[4];
#pragma unroll
  for (int kc = 0; kc < 4; kc++)
    qf[kc] = ldfragA(Q + (size_t)(qbase + l16) * DIMC + h * HDC + kc * 32 + half * 8);

  v8f o[8];
#pragma unroll
  for (int f = 0; f < 8; f++)
#pragma unroll
    for (int e = 0; e < 8; e++) o[f][e] = 0.f;
  float mi_[8], li_[8];
#pragma unroll
  for (int r = 0; r < 8; r++) { mi_[r] = -3.0e38f; li_[r] = 0.f; }

  const __bf16* Kh  = K + h * HDC;                 // rows: keys, stride DIMC
  const __bf16* VTh = VT + (size_t)(h * HDC) * Lk; // rows: d, stride Lk

#if USE_TDM
  const unsigned kt_base = (unsigned)(uintptr_t)(&Ktile[0][0][0]);
  const unsigned vt_base = (unsigned)(uintptr_t)(&Vtile[0][0][0]);
  if (wv == 0) {
    tdm_load_2d(Kh, kt_base, DIMC, Lk, DIMC, 128, 32);
    tdm_load_2d(VTh, vt_base, Lk, DIMC, Lk, 32, 128);
  }
  int par = 0;
#else
  const int par = 0;
#endif

  for (int j0 = 0; j0 < Lk; j0 += 32) {
#if USE_TDM
    if (wv == 0) {
      if (j0 + 32 < Lk) {
        tdm_load_2d(Kh + (size_t)(j0 + 32) * DIMC, kt_base + ((par ^ 1) << 13),
                    DIMC, Lk, DIMC, 128, 32);
        tdm_load_2d(VTh + j0 + 32, vt_base + ((par ^ 1) << 13),
                    Lk, DIMC, Lk, 32, 128);
        __builtin_amdgcn_s_wait_tensorcnt(2);   // current pair done
      } else {
        __builtin_amdgcn_s_wait_tensorcnt(0);
      }
    }
    __syncthreads();
#else
    __syncthreads();
    {
      // K tile: 32 keys x 128 d ; 256 threads x 16 bf16
      int kr = tid >> 3, kcol = (tid & 7) * 16;
      *(v16bf*)&Ktile[0][kr][kcol] =
          *(const v16bf*)(Kh + (size_t)(j0 + kr) * DIMC + kcol);
      // V tile: 128 d x 32 keys ; 256 threads x 16 bf16
      int vr = tid >> 1, vcol = (tid & 1) * 16;
      *(v16bf*)&Vtile[0][vr][vcol] =
          *(const v16bf*)(VTh + (size_t)vr * Lk + j0 + vcol);
    }
    __syncthreads();
#endif

    // S = Q * K^T for the 32 staged keys (two 16-key column tiles)
    v8f s[2];
#pragma unroll
    for (int t = 0; t < 2; t++) {
#pragma unroll
      for (int e = 0; e < 8; e++) s[t][e] = 0.f;
      const __bf16* krow = &Ktile[par][t * 16 + l16][0];
#pragma unroll
      for (int kc = 0; kc < 4; kc++) {
        v16bf kb = ldfragB(krow + kc * 32 + half * 8);
        s[t] = wmma_bf16(qf[kc], kb, s[t]);
      }
    }
    // online softmax
#pragma unroll
    for (int r = 0; r < 8; r++) {
      float v0 = s[0][r] * scale, v1 = s[1][r] * scale;
      float mx = fmaxf(v0, v1);
      for (int d = 1; d < 16; d <<= 1) mx = fmaxf(mx, __shfl_xor(mx, d, 32));
      float m_new = fmaxf(mi_[r], mx);
      float p0 = __expf(v0 - m_new), p1 = __expf(v1 - m_new);
      float ps = p0 + p1;
      for (int d = 1; d < 16; d <<= 1) ps += __shfl_xor(ps, d, 32);
      float corr = __expf(mi_[r] - m_new);
      li_[r] = li_[r] * corr + ps;
      mi_[r] = m_new;
#pragma unroll
      for (int f = 0; f < 8; f++) o[f][r] *= corr;
      P[wv][r + 8 * half][l16]      = f2bf(p0);
      P[wv][r + 8 * half][16 + l16] = f2bf(p1);
    }
    __syncthreads();
    v16bf pf;
#pragma unroll
    for (int i = 0; i < 8; i++) {
      pf[i]     = P[wv][l16][half * 8 + i];
      pf[8 + i] = P[wv][l16][16 + half * 8 + i];
    }
    // O += P * V (from staged V tile)
#pragma unroll
    for (int f = 0; f < 8; f++) {
      v16bf vb = ldfragB(&Vtile[par][f * 16 + l16][half * 8]);
      o[f] = wmma_bf16(pf, vb, o[f]);
    }
    __syncthreads();
#if USE_TDM
    par ^= 1;
#endif
  }

#pragma unroll
  for (int r = 0; r < 8; r++) {
    float invl = 1.f / li_[r];
    int row = qbase + r + 8 * half;
#pragma unroll
    for (int f = 0; f < 8; f++)
      Y[(size_t)row * DIMC + h * HDC + f * 16 + l16] = f2bf(o[f][r] * invl);
  }
}

// ---------------- workspace layout (bytes) ----------------
#define WQB 4718592ULL                 // 1536*1536*2 (bf16 weight)
#define WFB 27525120ULL                // 8960*1536*2
#define ABH 6291456ULL                 // 2048*1536*2 (bf16 activation)
#define AFH 12582912ULL                // 2048*1536*4 (fp32 activation)
#define O_EM   0ULL
#define O_COS  65536ULL
#define O_SIN  (O_COS + 524288ULL)
#define O_WSAQ 1114112ULL
#define O_WSAK (O_WSAQ + WQB)
#define O_WSAV (O_WSAK + WQB)
#define O_WSAO (O_WSAV + WQB)
#define O_WCAQ (O_WSAO + WQB)
#define O_WCAK (O_WCAQ + WQB)
#define O_WCAV (O_WCAK + WQB)
#define O_WCAO (O_WCAV + WQB)
#define O_WF1  (O_WCAO + WQB)
#define O_WF2  (O_WF1 + WFB)
#define O_XS   (O_WF2 + WFB)
#define O_QPB  (O_XS + ABH)
#define O_X1   (O_QPB + ABH)
#define O_X2   (O_X1 + AFH)
#define O_R    (O_X2 + AFH)
#define O_QB   O_R
#define O_KB   (O_R + ABH)
#define O_VT   (O_R + 2 * ABH)
#define O_Y    (O_R + 3 * ABH)
#define O_H    O_R
#define O_CTX  (O_R + 36700160ULL)

extern "C" void kernel_launch(void* const* d_in, const int* in_sizes, int n_in,
                              void* d_out, int out_size, void* d_ws, size_t ws_size,
                              hipStream_t stream) {
  const float* x          = (const float*)d_in[0];
  const float* e          = (const float*)d_in[1];
  const float* context    = (const float*)d_in[2];
  const float* freqs      = (const float*)d_in[3];
  const float* modulation = (const float*)d_in[4];
  const float* sa_qw = (const float*)d_in[5];  const float* sa_qb = (const float*)d_in[6];
  const float* sa_kw = (const float*)d_in[7];  const float* sa_kb = (const float*)d_in[8];
  const float* sa_vw = (const float*)d_in[9];  const float* sa_vb = (const float*)d_in[10];
  const float* sa_ow = (const float*)d_in[11]; const float* sa_ob = (const float*)d_in[12];
  const float* sa_nq = (const float*)d_in[13]; const float* sa_nk = (const float*)d_in[14];
  const float* ca_qw = (const float*)d_in[15]; const float* ca_qb = (const float*)d_in[16];
  const float* ca_kw = (const float*)d_in[17]; const float* ca_kb = (const float*)d_in[18];
  const float* ca_vw = (const float*)d_in[19]; const float* ca_vb = (const float*)d_in[20];
  const float* ca_ow = (const float*)d_in[21]; const float* ca_ob = (const float*)d_in[22];
  const float* ca_nq = (const float*)d_in[23]; const float* ca_nk = (const float*)d_in[24];
  const float* n3_w  = (const float*)d_in[25]; const float* n3_b  = (const float*)d_in[26];
  const float* f_w1  = (const float*)d_in[27]; const float* f_b1  = (const float*)d_in[28];
  const float* f_w2  = (const float*)d_in[29]; const float* f_b2  = (const float*)d_in[30];

  char* ws = (char*)d_ws;
  float*  EM   = (float*)(ws + O_EM);
  float*  cosT = (float*)(ws + O_COS);
  float*  sinT = (float*)(ws + O_SIN);
  __bf16* WSAQ = (__bf16*)(ws + O_WSAQ);
  __bf16* WSAK = (__bf16*)(ws + O_WSAK);
  __bf16* WSAV = (__bf16*)(ws + O_WSAV);
  __bf16* WSAO = (__bf16*)(ws + O_WSAO);
  __bf16* WCAQ = (__bf16*)(ws + O_WCAQ);
  __bf16* WCAK = (__bf16*)(ws + O_WCAK);
  __bf16* WCAV = (__bf16*)(ws + O_WCAV);
  __bf16* WCAO = (__bf16*)(ws + O_WCAO);
  __bf16* WF1  = (__bf16*)(ws + O_WF1);
  __bf16* WF2  = (__bf16*)(ws + O_WF2);
  __bf16* XSb  = (__bf16*)(ws + O_XS);
  __bf16* QPb  = (__bf16*)(ws + O_QPB);
  float*  X1   = (float*)(ws + O_X1);
  float*  X2   = (float*)(ws + O_X2);
  __bf16* QB   = (__bf16*)(ws + O_QB);
  __bf16* KB   = (__bf16*)(ws + O_KB);
  __bf16* VT   = (__bf16*)(ws + O_VT);
  __bf16* Yb   = (__bf16*)(ws + O_Y);
  __bf16* Hbf  = (__bf16*)(ws + O_H);
  __bf16* CTXB = (__bf16*)(ws + O_CTX);
  float*  OUT  = (float*)d_out;

  dim3 g6x32(6, 32), g6x8(6, 8), g35x32(35, 32);
  const int NW8 = DIMC * DIMC / 8 / 256;
  const int NF8 = FFNC * DIMC / 8 / 256;
  const int NC8 = LKXC * DIMC / 8 / 256;

  // ---- per-launch conversions: weights + context -> bf16 ----
  cvt_k<<<NW8, 256, 0, stream>>>(sa_qw, WSAQ, DIMC * DIMC / 8);
  cvt_k<<<NW8, 256, 0, stream>>>(sa_kw, WSAK, DIMC * DIMC / 8);
  cvt_k<<<NW8, 256, 0, stream>>>(sa_vw, WSAV, DIMC * DIMC / 8);
  cvt_k<<<NW8, 256, 0, stream>>>(sa_ow, WSAO, DIMC * DIMC / 8);
  cvt_k<<<NW8, 256, 0, stream>>>(ca_qw, WCAQ, DIMC * DIMC / 8);
  cvt_k<<<NW8, 256, 0, stream>>>(ca_kw, WCAK, DIMC * DIMC / 8);
  cvt_k<<<NW8, 256, 0, stream>>>(ca_vw, WCAV, DIMC * DIMC / 8);
  cvt_k<<<NW8, 256, 0, stream>>>(ca_ow, WCAO, DIMC * DIMC / 8);
  cvt_k<<<NF8, 256, 0, stream>>>(f_w1, WF1, FFNC * DIMC / 8);
  cvt_k<<<NF8, 256, 0, stream>>>(f_w2, WF2, FFNC * DIMC / 8);
  cvt_k<<<NC8, 256, 0, stream>>>(context, CTXB, LKXC * DIMC / 8);

  prep_k<<<512, 256, 0, stream>>>(e, modulation, freqs, EM, cosT, sinT);

  // ---- self attention ----
  ln_k<<<LQC, 256, 0, stream>>>(x, EM + 1 * DIMC, EM + 0 * DIMC, XSb, 0);

  gemm_k<EPI_BF16><<<g6x32, 256, 0, stream>>>(XSb, WSAQ, sa_qb, nullptr, nullptr,
                                              nullptr, QPb, LQC, DIMC, DIMC);
  rmsrope_k<<<LQC * NHC, 128, 0, stream>>>(QPb, sa_nq, cosT, sinT, QB, 1);

  gemm_k<EPI_BF16><<<g6x32, 256, 0, stream>>>(XSb, WSAK, sa_kb, nullptr, nullptr,
                                              nullptr, QPb, LQC, DIMC, DIMC);
  rmsrope_k<<<LQC * NHC, 128, 0, stream>>>(QPb, sa_nk, cosT, sinT, KB, 1);

  gemm_k<EPI_BF16><<<g6x32, 256, 0, stream>>>(XSb, WSAV, sa_vb, nullptr, nullptr,
                                              nullptr, QPb, LQC, DIMC, DIMC);
  transpose_k<<<(LQC * DIMC + 255) / 256, 256, 0, stream>>>(QPb, VT, LQC);

  attn_k<<<dim3(LQC / 128, NHC), 256, 0, stream>>>(QB, KB, VT, Yb, LQC, SCALEC);

  gemm_k<EPI_RESID_SCALE><<<g6x32, 256, 0, stream>>>(Yb, WSAO, sa_ob, x,
                                                     EM + 2 * DIMC, X1, nullptr,
                                                     LQC, DIMC, DIMC);

  // ---- cross attention ----
  ln_k<<<LQC, 256, 0, stream>>>(X1, n3_w, n3_b, XSb, 1);

  gemm_k<EPI_BF16><<<g6x32, 256, 0, stream>>>(XSb, WCAQ, ca_qb, nullptr, nullptr,
                                              nullptr, QPb, LQC, DIMC, DIMC);
  rmsrope_k<<<LQC * NHC, 128, 0, stream>>>(QPb, ca_nq, cosT, sinT, QB, 0);

  gemm_k<EPI_BF16><<<g6x8, 256, 0, stream>>>(CTXB, WCAK, ca_kb, nullptr, nullptr,
                                             nullptr, QPb, LKXC, DIMC, DIMC);
  rmsrope_k<<<LKXC * NHC, 128, 0, stream>>>(QPb, ca_nk, cosT, sinT, KB, 0);

  gemm_k<EPI_BF16><<<g6x8, 256, 0, stream>>>(CTXB, WCAV, ca_vb, nullptr, nullptr,
                                             nullptr, QPb, LKXC, DIMC, DIMC);
  transpose_k<<<(LKXC * DIMC + 255) / 256, 256, 0, stream>>>(QPb, VT, LKXC);

  attn_k<<<dim3(LQC / 128, NHC), 256, 0, stream>>>(QB, KB, VT, Yb, LKXC, SCALEC);

  gemm_k<EPI_RESID><<<g6x32, 256, 0, stream>>>(Yb, WCAO, ca_ob, X1, nullptr,
                                               X2, nullptr, LQC, DIMC, DIMC);

  // ---- FFN ----
  ln_k<<<LQC, 256, 0, stream>>>(X2, EM + 4 * DIMC, EM + 3 * DIMC, XSb, 0);

  gemm_k<EPI_GELU_BF16><<<g35x32, 256, 0, stream>>>(XSb, WF1, f_b1, nullptr, nullptr,
                                                    nullptr, Hbf, LQC, FFNC, DIMC);

  gemm_k<EPI_RESID_SCALE><<<g6x32, 256, 0, stream>>>(Hbf, WF2, f_b2, X2,
                                                     EM + 5 * DIMC, OUT, nullptr,
                                                     LQC, DIMC, FFNC);
}
